// GraphTransformerLayer_40157944218454
// MI455X (gfx1250) — matrix operand order repaired and checked by reading, not compiled, not run on hardware
//
#include <hip/hip_runtime.h>

// ---------------------------------------------------------------------------
// CDNA5 / gfx1250 transformer encoder layer, bf16 WMMA everywhere.
// All fragments are assembled from K-contiguous storage with 16B vector loads:
//   A-layout frag = two 16B chunks at K = 8*kh and 16+8*kh
//   B-layout frag = two 16B chunks at K = 8*kh and 16+8*kh
// Weights are pre-transposed to [N][K] bf16; V is written pre-transposed
// [B,H,dk,L] by the QKV GEMM epilogue so flash attention needs no transposes.
// GEMM uses double-buffered LDS tiles + global_prefetch of the k+2 tile.
// attn_bias (256 MB > 192 MB L2) is streamed with non-temporal loads.
// ---------------------------------------------------------------------------

typedef __attribute__((ext_vector_type(16))) __bf16 bf16x16;
typedef __attribute__((ext_vector_type(8)))  float  floatx8;

union AFrag { bf16x16 v; unsigned short u[16]; uint4 q[2]; };
union CFrag { floatx8 v; float f[8]; };

__device__ __forceinline__ unsigned short f32_to_bf16(float f) {
  union { float f; unsigned int u; } c; c.f = f;
  unsigned int u = c.u;
  if ((u & 0x7fffffffu) > 0x7f800000u)               // NaN -> quiet NaN
    return (unsigned short)((u >> 16) | 0x0040u);
  unsigned int r = u + 0x7fffu + ((u >> 16) & 1u);   // round-to-nearest-even
  return (unsigned short)(r >> 16);
}

// A-layout fragment from a K-major row (row points at K=0 of this row)
__device__ __forceinline__ void load_a_frag(AFrag& f, const unsigned short* row, int kh) {
  f.q[0] = *(const uint4*)(row + 8 * kh);        // K = 8kh .. 8kh+7
  f.q[1] = *(const uint4*)(row + 16 + 8 * kh);   // K = 16+8kh .. 16+8kh+7
}
// B-layout fragment from a K-major row (lane's column row)
__device__ __forceinline__ void load_b_frag(AFrag& f, const unsigned short* row, int kh) {
  f.q[0] = *(const uint4*)(row + 8 * kh);        // K = 8kh .. 8kh+7
  f.q[1] = *(const uint4*)(row + 16 + 8 * kh);   // K = 16+8kh .. 16+8kh+7
}

// ---------------------------------------------------------------------------
// f32 -> bf16 conversions (plain and transposing, for weights -> [N][K])
// ---------------------------------------------------------------------------
__global__ void cvt_bf16_kernel(const float* __restrict__ src,
                                unsigned short* __restrict__ dst, int n) {
  int i = blockIdx.x * 256 + threadIdx.x;
  if (i < n) dst[i] = f32_to_bf16(src[i]);
}

__global__ void cvt_bf16_t_kernel(const float* __restrict__ src,  // [K,N]
                                  unsigned short* __restrict__ dst, // [N,K]
                                  int K, int N) {
  int i = blockIdx.x * 256 + threadIdx.x;
  if (i < K * N) {
    int n = i / K, k = i - n * K;
    dst[i] = f32_to_bf16(src[(size_t)k * N + n]);
  }
}

// ---------------------------------------------------------------------------
// Tiled bf16 WMMA GEMM: C[M,N] = A[M,K] * Bt[N,K]^T (+bias) (+relu)
// Block tile 128x128, 8 waves; wave tile 32x64 (8 WMMA per K-step).
// Double-buffered LDS; global_prefetch_b8 of the k+2 tile rows.
// Optional transposed-V output: columns n >= vtSplit go to vT[B,H,dk,L].
// M,N multiples of 128; K multiple of 32.
// ---------------------------------------------------------------------------
__global__ __launch_bounds__(256)
void gemm_bf16_wmma(const unsigned short* __restrict__ A,
                    const unsigned short* __restrict__ Bt,
                    const float* __restrict__ bias,
                    float* __restrict__ outF,
                    unsigned short* __restrict__ outBf,
                    unsigned short* __restrict__ outVT,
                    int M, int N, int K, int relu, int vtSplit) {
  __shared__ unsigned short lA[2][128 * 32];   // [m][k]
  __shared__ unsigned short lB[2][128 * 32];   // [n][k]
  const int tid  = threadIdx.x;
  const int lane = tid & 31;
  const int wave = tid >> 5;
  const int m0 = blockIdx.y * 128;
  const int n0 = blockIdx.x * 128;
  const int wm = (wave >> 1) * 32;   // 0,32,64,96
  const int wn = (wave & 1) * 64;    // 0,64
  const int ml = lane & 15;
  const int kh = lane >> 4;

  CFrag acc[2][4];
#pragma unroll
  for (int mi = 0; mi < 2; ++mi)
#pragma unroll
    for (int ni = 0; ni < 4; ++ni)
#pragma unroll
      for (int j = 0; j < 8; ++j) acc[mi][ni].f[j] = 0.f;

  auto stage = [&](int buf, int k0) {
#pragma unroll
    for (int c = tid; c < 512; c += 256) {
      int r = c >> 2, o = (c & 3) * 8;
      *(uint4*)&lA[buf][r * 32 + o] =
          *(const uint4*)&A[(size_t)(m0 + r) * K + k0 + o];
      *(uint4*)&lB[buf][r * 32 + o] =
          *(const uint4*)&Bt[(size_t)(n0 + r) * K + k0 + o];
    }
  };

  const int nk = K / 32;
  stage(0, 0);
  __syncthreads();

  for (int ki = 0; ki < nk; ++ki) {
    const int buf = ki & 1;
    if (ki + 1 < nk) stage(buf ^ 1, (ki + 1) * 32);
    if (ki + 2 < nk) {
      // pull the tile-after-next toward the caches (64B per row)
      int kp = (ki + 2) * 32;
      if (tid < 128) __builtin_prefetch(&A[(size_t)(m0 + tid) * K + kp], 0, 1);
      else           __builtin_prefetch(&Bt[(size_t)(n0 + tid - 128) * K + kp], 0, 1);
    }

    AFrag af[2];
#pragma unroll
    for (int mi = 0; mi < 2; ++mi)
      load_a_frag(af[mi], &lA[buf][(wm + mi * 16 + ml) * 32], kh);

#pragma unroll
    for (int ni = 0; ni < 4; ++ni) {
      AFrag bf;
      load_b_frag(bf, &lB[buf][(wn + ni * 16 + ml) * 32], kh);
#pragma unroll
      for (int mi = 0; mi < 2; ++mi)
        acc[mi][ni].v = __builtin_amdgcn_wmma_f32_16x16x32_bf16(
            false, af[mi].v, false, bf.v, (short)0, acc[mi][ni].v, false, false);
    }
    __syncthreads();
  }

#pragma unroll
  for (int mi = 0; mi < 2; ++mi)
#pragma unroll
    for (int ni = 0; ni < 4; ++ni) {
      int n  = n0 + wn + ni * 16 + ml;
      int mb = m0 + wm + mi * 16 + kh * 8;
      float bv = bias ? bias[n] : 0.f;
      if (outVT && n >= vtSplit) {
        // transposed V write: fixed (head, d), 8 consecutive keys -> one b128
        int dAll = n - vtSplit;
        int hh = dAll >> 6, dd = dAll & 63;
        int b = mb >> 10, key0 = mb & 1023;
        union { uint4 q; unsigned short u[8]; } pk;
#pragma unroll
        for (int j = 0; j < 8; ++j)
          pk.u[j] = f32_to_bf16(acc[mi][ni].f[j] + bv);
        *(uint4*)&outVT[(((size_t)b * 8 + hh) * 64 + dd) * 1024 + key0] = pk.q;
      } else {
#pragma unroll
        for (int j = 0; j < 8; ++j) {
          float v = acc[mi][ni].f[j] + bv;
          if (relu) v = fmaxf(v, 0.f);
          size_t idx = (size_t)(mb + j) * N + n;
          if (outF)  outF[idx]  = v;
          if (outBf) outBf[idx] = f32_to_bf16(v);
        }
      }
    }
}

// ---------------------------------------------------------------------------
// Flash attention: one wave per (b, head, 16-query tile).
// qkv bf16 [B,L,1536] (q|k|-), vT bf16 [B,H,64,L]. Online softmax, bias+mask.
// ---------------------------------------------------------------------------
__global__ __launch_bounds__(32)
void attn_flash_kernel(const unsigned short* __restrict__ qkv,
                       const unsigned short* __restrict__ vT,
                       const float* __restrict__ bias,
                       const int* __restrict__ mask,
                       unsigned short* __restrict__ attnOut) {
  const int L = 1024, D3 = 1536, HD = 64;
  __shared__ unsigned short lpu[16 * 32];   // P tile, bf16 [m][key]
  const int lane = (int)threadIdx.x;
  const int qb = blockIdx.x, hh = blockIdx.y, b = blockIdx.z;
  const int qBase = qb * 16;
  const int ml = lane & 15;
  const int hf = lane >> 4;

  // Q as two A-fragments (dk=64), direct 16B global loads
  AFrag qa[2];
  {
    const unsigned short* qrow =
        qkv + (size_t)(b * L + qBase + ml) * D3 + hh * HD;
    load_a_frag(qa[0], qrow, hf);
    load_a_frag(qa[1], qrow + 32, hf);
  }

  CFrag o[4];
#pragma unroll
  for (int t = 0; t < 4; ++t)
#pragma unroll
    for (int j = 0; j < 8; ++j) o[t].f[j] = 0.f;
  float rowMax[8], rowSum[8];
#pragma unroll
  for (int j = 0; j < 8; ++j) { rowMax[j] = -3.0e38f; rowSum[j] = 0.f; }

  for (int kb = 0; kb < L / 32; ++kb) {
    const int keyBase = kb * 32;

    // prefetch next key-block streams (bias rows + K rows)
    if (kb + 1 < L / 32) {
      int nb = keyBase + 32;
      if (hf == 0) {
        size_t bb = (((size_t)(b * 8 + hh)) * L + qBase + ml) * L;
        __builtin_prefetch(&bias[bb + nb], 0, 1);
      } else {
        const unsigned short* kr =
            qkv + (size_t)(b * L + nb + ml * 2) * D3 + 512 + hh * HD;
        __builtin_prefetch(kr, 0, 1);
        __builtin_prefetch(kr + D3, 0, 1);
      }
    }

    // S = Q K^T for 32 keys (4 WMMAs); K fragments are dk-contiguous 16B loads
    CFrag s[2];
#pragma unroll
    for (int t = 0; t < 2; ++t) {
      AFrag bk0, bk1;
      const unsigned short* krow =
          qkv + (size_t)(b * L + keyBase + t * 16 + ml) * D3 + 512 + hh * HD;
      load_b_frag(bk0, krow, hf);
      load_b_frag(bk1, krow + 32, hf);
#pragma unroll
      for (int j = 0; j < 8; ++j) s[t].f[j] = 0.f;
      s[t].v = __builtin_amdgcn_wmma_f32_16x16x32_bf16(
          false, qa[0].v, false, bk0.v, (short)0, s[t].v, false, false);
      s[t].v = __builtin_amdgcn_wmma_f32_16x16x32_bf16(
          false, qa[1].v, false, bk1.v, (short)0, s[t].v, false, false);
    }

    // bias + mask + online softmax (this half-wave owns rows hf*8 + j)
    float p0[8], p1[8];
#pragma unroll
    for (int j = 0; j < 8; ++j) {
      int m = qBase + hf * 8 + j;
      int key0 = keyBase + ml, key1 = keyBase + 16 + ml;
      size_t bb = (((size_t)(b * 8 + hh)) * L + m) * L;
      float v0 = s[0].f[j] * 0.125f + __builtin_nontemporal_load(&bias[bb + key0]);
      float v1 = s[1].f[j] * 0.125f + __builtin_nontemporal_load(&bias[bb + key1]);
      if (mask[b * L + key0] == 0) v0 = -3.0e38f;
      if (mask[b * L + key1] == 0) v1 = -3.0e38f;
      float lm = fmaxf(v0, v1);
#pragma unroll
      for (int x = 1; x <= 8; x <<= 1) lm = fmaxf(lm, __shfl_xor(lm, x, 32));
      float nmax = fmaxf(rowMax[j], lm);
      float alpha = __expf(rowMax[j] - nmax);
      float e0 = __expf(v0 - nmax), e1 = __expf(v1 - nmax);
      p0[j] = e0; p1[j] = e1;
      float ls = e0 + e1;
#pragma unroll
      for (int x = 1; x <= 8; x <<= 1) ls += __shfl_xor(ls, x, 32);
      rowSum[j] = rowSum[j] * alpha + ls;
      rowMax[j] = nmax;
#pragma unroll
      for (int t = 0; t < 4; ++t) o[t].f[j] *= alpha;
    }

    // C-layout P -> A-layout bf16 fragment via small LDS tile
    __syncthreads();
#pragma unroll
    for (int j = 0; j < 8; ++j) {
      lpu[(hf * 8 + j) * 32 + ml]      = f32_to_bf16(p0[j]);
      lpu[(hf * 8 + j) * 32 + 16 + ml] = f32_to_bf16(p1[j]);
    }
    __syncthreads();
    AFrag pa;
    load_a_frag(pa, &lpu[ml * 32], hf);

    // O += P V (4 WMMAs); V fragments are key-contiguous 16B loads from vT
#pragma unroll
    for (int t = 0; t < 4; ++t) {
      AFrag vb;
      const unsigned short* vrow =
          vT + (((size_t)(b * 8 + hh)) * 64 + t * 16 + ml) * 1024 + keyBase;
      load_b_frag(vb, vrow, hf);
      o[t].v = __builtin_amdgcn_wmma_f32_16x16x32_bf16(
          false, pa.v, false, vb.v, (short)0, o[t].v, false, false);
    }
  }

  // normalize and write [B,L,512] bf16 (heads re-interleaved)
#pragma unroll
  for (int j = 0; j < 8; ++j) {
    float inv = rowSum[j] > 0.f ? 1.f / rowSum[j] : 0.f;
    int row = qBase + hf * 8 + j;
#pragma unroll
    for (int t = 0; t < 4; ++t)
      attnOut[(size_t)(b * L + row) * 512 + hh * HD + t * 16 + ml] =
          f32_to_bf16(o[t].f[j] * inv);
  }
}

// ---------------------------------------------------------------------------
// Residual add + LayerNorm over D=512, float4 vectorized. One block per row.
// ---------------------------------------------------------------------------
__global__ __launch_bounds__(128)
void ln_kernel(const float* __restrict__ xa, const float* __restrict__ xb,
               const float* __restrict__ gamma, const float* __restrict__ beta,
               float* __restrict__ outF, unsigned short* __restrict__ outBf) {
  const int D = 512;
  __shared__ float red[128];
  const int row = blockIdx.x;
  const int tid = threadIdx.x;
  const int c0 = tid * 4;
  float4 a = *(const float4*)&xa[(size_t)row * D + c0];
  float4 bq = *(const float4*)&xb[(size_t)row * D + c0];
  float v[4] = {a.x + bq.x, a.y + bq.y, a.z + bq.z, a.w + bq.w};
  float s = v[0] + v[1] + v[2] + v[3];
  red[tid] = s; __syncthreads();
  for (int off = 64; off > 0; off >>= 1) {
    if (tid < off) red[tid] += red[tid + off];
    __syncthreads();
  }
  float mu = red[0] / D;
  __syncthreads();
  float s2 = 0.f;
#pragma unroll
  for (int i = 0; i < 4; ++i) { float d = v[i] - mu; s2 += d * d; }
  red[tid] = s2; __syncthreads();
  for (int off = 64; off > 0; off >>= 1) {
    if (tid < off) red[tid] += red[tid + off];
    __syncthreads();
  }
  float rstd = rsqrtf(red[0] / D + 1e-5f);
  float4 g = *(const float4*)&gamma[c0];
  float4 be = *(const float4*)&beta[c0];
  float y0 = (v[0] - mu) * rstd * g.x + be.x;
  float y1 = (v[1] - mu) * rstd * g.y + be.y;
  float y2 = (v[2] - mu) * rstd * g.z + be.z;
  float y3 = (v[3] - mu) * rstd * g.w + be.w;
  *(float4*)&outF[(size_t)row * D + c0] = make_float4(y0, y1, y2, y3);
  if (outBf) {
    union { uint2 q; unsigned short u[4]; } pk;
    pk.u[0] = f32_to_bf16(y0); pk.u[1] = f32_to_bf16(y1);
    pk.u[2] = f32_to_bf16(y2); pk.u[3] = f32_to_bf16(y3);
    *(uint2*)&outBf[(size_t)row * D + c0] = pk.q;
  }
}

// ---------------------------------------------------------------------------
extern "C" void kernel_launch(void* const* d_in, const int* in_sizes, int n_in,
                              void* d_out, int out_size, void* d_ws, size_t ws_size,
                              hipStream_t stream) {
  (void)in_sizes; (void)n_in; (void)out_size; (void)ws_size;
  const float* h    = (const float*)d_in[0];
  const float* bias = (const float*)d_in[1];
  const int*   mask = (const int*)d_in[2];
  const float* Wqkv = (const float*)d_in[3];
  const float* bqkv = (const float*)d_in[4];
  const float* Wo   = (const float*)d_in[5];
  const float* bo   = (const float*)d_in[6];
  const float* W1   = (const float*)d_in[7];
  const float* b1   = (const float*)d_in[8];
  const float* W2   = (const float*)d_in[9];
  const float* b2   = (const float*)d_in[10];
  const float* g1   = (const float*)d_in[11];
  const float* be1  = (const float*)d_in[12];
  const float* g2   = (const float*)d_in[13];
  const float* be2  = (const float*)d_in[14];

  const int B = 8, L = 1024, D = 512, FF = 2048;
  const int M = B * L;                         // 8192 rows
  const int BIG = 1 << 30;

  char* ws = (char*)d_ws;
  size_t off = 0;
  auto carve = [&](size_t bytes) -> char* {
    char* p = ws + off; off += (bytes + 255) & ~(size_t)255; return p;
  };
  unsigned short* hbf    = (unsigned short*)carve((size_t)M * D * 2);
  unsigned short* wqkvT  = (unsigned short*)carve((size_t)D * 3 * D * 2);
  unsigned short* woT    = (unsigned short*)carve((size_t)D * D * 2);
  unsigned short* w1T    = (unsigned short*)carve((size_t)D * FF * 2);
  unsigned short* w2T    = (unsigned short*)carve((size_t)FF * D * 2);
  unsigned short* qkvbf  = (unsigned short*)carve((size_t)M * 3 * D * 2);
  unsigned short* vTbuf  = (unsigned short*)carve((size_t)M * D * 2);
  unsigned short* attnbf = (unsigned short*)carve((size_t)M * D * 2);
  float*          projf  = (float*)carve((size_t)M * D * 4);   // Wo out / FFN out
  float*          h1f    = (float*)carve((size_t)M * D * 4);
  unsigned short* h1bf   = (unsigned short*)carve((size_t)M * D * 2);
  unsigned short* ff1bf  = (unsigned short*)carve((size_t)M * FF * 2);

  // conversions: activations plain, weights transposed to [N][K]
  cvt_bf16_kernel<<<(M * D + 255) / 256, 256, 0, stream>>>(h, hbf, M * D);
  cvt_bf16_t_kernel<<<(D * 3 * D + 255) / 256, 256, 0, stream>>>(Wqkv, wqkvT, D, 3 * D);
  cvt_bf16_t_kernel<<<(D * D + 255) / 256, 256, 0, stream>>>(Wo, woT, D, D);
  cvt_bf16_t_kernel<<<(D * FF + 255) / 256, 256, 0, stream>>>(W1, w1T, D, FF);
  cvt_bf16_t_kernel<<<(FF * D + 255) / 256, 256, 0, stream>>>(W2, w2T, FF, D);

  // QKV projection; V columns (>=1024) written pre-transposed to vTbuf
  gemm_bf16_wmma<<<dim3((3 * D) / 128, M / 128), 256, 0, stream>>>(
      hbf, wqkvT, bqkv, nullptr, qkvbf, vTbuf, M, 3 * D, D, 0, 1024);

  // flash attention per (b, head, 16-query tile)
  attn_flash_kernel<<<dim3(L / 16, 8, B), 32, 0, stream>>>(
      qkvbf, vTbuf, bias, mask, attnbf);

  // output projection: attn x Wo + bo -> f32
  gemm_bf16_wmma<<<dim3(D / 128, M / 128), 256, 0, stream>>>(
      attnbf, woT, bo, projf, nullptr, nullptr, M, D, D, 0, BIG);

  // h1 = LN(h + proj)
  ln_kernel<<<M, 128, 0, stream>>>(h, projf, g1, be1, h1f, h1bf);

  // ff1 = relu(h1 x W1 + b1) -> bf16
  gemm_bf16_wmma<<<dim3(FF / 128, M / 128), 256, 0, stream>>>(
      h1bf, w1T, b1, nullptr, ff1bf, nullptr, M, FF, D, 1, BIG);

  // ff2 = ff1 x W2 + b2 -> f32 (reuse projf)
  gemm_bf16_wmma<<<dim3(D / 128, M / 128), 256, 0, stream>>>(
      ff1bf, w2T, b2, projf, nullptr, nullptr, M, D, FF, 0, BIG);

  // out = LN(h1 + ff2)
  ln_kernel<<<M, 128, 0, stream>>>(h1f, projf, g2, be2, (float*)d_out, nullptr);
}
